// PAModule_16277926052512
// MI455X (gfx1250) — compile-verified, matrix operand order checked by reading, not run on hardware
//
#include <hip/hip_runtime.h>
#include <hip/hip_bf16.h>

typedef __attribute__((ext_vector_type(16))) _Float16 v16h;
typedef __attribute__((ext_vector_type(8)))  float    v8f;
typedef __attribute__((ext_vector_type(4)))  unsigned int u32x4;
typedef __attribute__((ext_vector_type(8)))  unsigned int u32x8;

#define Bn 4
#define Cc 512
#define CQ 64
#define Nn 4096
#define NCH (Nn / 64)   // 64-key chunks

union Frag16 { unsigned int u[8]; v16h h; };

// 16-bit A/B matrix VGPR->K mapping (CDNA5 ISA 7.12.2):
// VGPR v holds K pair starting at: v<4 ? 2v+8*half : 16+2(v-4)+8*half
__device__ __forceinline__ int kpair(int v, int half) {
  return (v < 4) ? (2 * v + 8 * half) : (16 + 2 * (v - 4) + 8 * half);
}

// ---------------------------------------------------------------------------
// Tensor Data Mover: 2-D tile load Global->LDS (CDNA5 ISA ch.8, D# groups 0/1).
// 16-bit elements. One descriptor per call; tracked by TENSORcnt.
//   lds_off   : LDS byte address of tile start
//   gptr      : global address of tile start (element [0,0])
//   tile_d0/1 : tile size (elements / rows)
//   tens_d0/1 : tensor extents for OOB checks (elements / rows)
//   stride0   : row stride in elements
// ---------------------------------------------------------------------------
__device__ __forceinline__ void tdm_load_2d_f16(unsigned lds_off, const void* gptr,
                                                unsigned tile_d0, unsigned tile_d1,
                                                unsigned tens_d0, unsigned tens_d1,
                                                unsigned stride0)
{
  unsigned long long ga = (unsigned long long)(size_t)gptr;
  u32x4 g0 = {
    1u,                                                   // count=1, user D#
    lds_off,                                              // lds_addr
    (unsigned)ga,                                         // global_addr[31:0]
    (unsigned)((ga >> 32) & 0x01FFFFFFu) | 0x80000000u    // addr[56:32] | type=2
  };
  u32x8 g1 = {
    0x00010000u,                                          // data_size=1 (2B), mask=0
    (tens_d0 & 0xFFFFu) << 16,                            // tensor_dim0[15:0]
    ((tens_d1 & 0xFFFFu) << 16) | ((tens_d0 >> 16) & 0xFFFFu),
    ((tile_d0 & 0xFFFFu) << 16) | ((tens_d1 >> 16) & 0xFFFFu),
    (tile_d1 & 0xFFFFu),                                  // tile_dim1, tile_dim2=0
    stride0,                                              // tensor_dim0_stride lo32
    0u, 0u                                                // stride hi / dim1_stride
  };
  asm volatile("tensor_load_to_lds %0, %1" :: "s"(g0), "s"(g1) : "memory");
}

__device__ __forceinline__ void wait_tensorcnt_le(int n) {
#if __has_builtin(__builtin_amdgcn_s_wait_tensorcnt)
  if (n == 0) __builtin_amdgcn_s_wait_tensorcnt(0);
  else        __builtin_amdgcn_s_wait_tensorcnt(1);
#else
  if (n == 0) asm volatile("s_wait_tensorcnt 0x0" ::: "memory");
  else        asm volatile("s_wait_tensorcnt 0x1" ::: "memory");
#endif
}

// ---------------------------------------------------------------------------
// Kernel 0: one-time weight prep. Unify wq/wk/wv into f16 Wh[640][512] and
// biases into biasc[640] so the projection GEMM does zero f32->f16 work.
// ---------------------------------------------------------------------------
__global__ __launch_bounds__(256)
void prep_w_kernel(const float* __restrict__ wq, const float* __restrict__ bq,
                   const float* __restrict__ wk, const float* __restrict__ bk,
                   const float* __restrict__ wvm, const float* __restrict__ bv,
                   _Float16* __restrict__ Wh, float* __restrict__ biasc)
{
  int idx = blockIdx.x * 256 + threadIdx.x;   // exactly 640*512 threads
  int o = idx >> 9, c = idx & 511;
  float v;
  if (o < 64)        v = wq[o * Cc + c];
  else if (o < 128)  v = wk[(o - 64) * Cc + c];
  else               v = wvm[(o - 128) * Cc + c];
  Wh[idx] = (_Float16)v;
  if (idx < 640) {
    float bb = (idx < 64) ? bq[idx] : (idx < 128 ? bk[idx - 64] : bv[idx - 128]);
    biasc[idx] = bb;
  }
}

// ---------------------------------------------------------------------------
// Kernel 1: fused QKV projection.  QKV[o,n] = sum_c Wh[o,c]*x[b,c,n] + bias[o]
//   o <  64 -> Qb[b][n][o]      o < 128 -> Kb[b][n][o-64]      else Vt[b][o-128][n]
// Block: 256 thr (8 waves) = 64 spatial rows x 128 output channels.
// ---------------------------------------------------------------------------
__global__ __launch_bounds__(256)
void proj_kernel(const float* __restrict__ x,
                 const _Float16* __restrict__ Wh, const float* __restrict__ biasc,
                 _Float16* __restrict__ Qb, _Float16* __restrict__ Kb,
                 _Float16* __restrict__ Vt)
{
  const int og   = blockIdx.x;            // 0..4 : 128 output channels
  const int nt   = blockIdx.y;            // 0..63: 64 spatial rows
  const int b    = blockIdx.z;
  const int tid  = threadIdx.x;
  const int lane = tid & 31;
  const int wvid = tid >> 5;
  const int half = lane >> 4;
  const int l15  = lane & 15;
  const int nbase = nt * 64;
  const int ocol  = og * 128 + wvid * 16 + l15;   // this lane's output channel

  __shared__ _Float16 sx[64 * 32];        // x^T tile [n][c]

  const _Float16* wrow = Wh + (size_t)ocol * Cc;
  const float bias = biasc[ocol];

  v8f acc[4] = {};

  for (int cb = 0; cb < Cc; cb += 32) {
    for (int i = tid; i < 64 * 32; i += 256) {
      int n = i & 63, c = i >> 6;
      float val = x[((size_t)(b * Cc + cb + c)) * Nn + nbase + n];
      sx[n * 32 + c] = (_Float16)val;
    }
    __syncthreads();

    Frag16 bf;
    #pragma unroll
    for (int v = 0; v < 8; ++v) {
      int k0 = kpair(v, half);
      bf.u[v] = *(const unsigned int*)&wrow[cb + k0];
    }
    #pragma unroll
    for (int rt = 0; rt < 4; ++rt) {
      Frag16 af;
      int m = rt * 16 + l15;
      #pragma unroll
      for (int v = 0; v < 8; ++v) {
        int k0 = kpair(v, half);
        af.u[v] = *(const unsigned int*)&sx[m * 32 + k0];
      }
      acc[rt] = __builtin_amdgcn_wmma_f32_16x16x32_f16(
          false, af.h, false, bf.h, (short)0, acc[rt], false, false);
    }
    __syncthreads();
  }

  // resolve destination once: dst[n * rstride]
  _Float16* dst; int rstride;
  if (ocol < 64)       { dst = Qb + (size_t)b * Nn * CQ + ocol;             rstride = CQ; }
  else if (ocol < 128) { dst = Kb + (size_t)b * Nn * CQ + (ocol - 64);      rstride = CQ; }
  else                 { dst = Vt + ((size_t)(b * Cc + (ocol - 128))) * Nn; rstride = 1;  }

  #pragma unroll
  for (int rt = 0; rt < 4; ++rt)
    #pragma unroll
    for (int j = 0; j < 8; ++j) {
      int n = nbase + rt * 16 + j + 8 * half;
      dst[(size_t)n * rstride] = (_Float16)(acc[rt][j] + bias);
    }
}

// ---------------------------------------------------------------------------
// Kernel 2: softmax row statistics (row max, then sum of exp) over S = Q K^T.
// Block: 64 query rows; wave w: row-subtile w%4, key col-tiles {2*(w/4)+0,1}.
// ---------------------------------------------------------------------------
__global__ __launch_bounds__(256)
void stats_kernel(const _Float16* __restrict__ Qb, const _Float16* __restrict__ Kb,
                  float* __restrict__ gmax, float* __restrict__ gsum)
{
  const int nt = blockIdx.x, b = blockIdx.y;
  const int tid = threadIdx.x, lane = tid & 31, wvid = tid >> 5;
  const int half = lane >> 4, l15 = lane & 15;
  const int nbase = nt * 64;
  const int rt = wvid & 3;
  const int cp = wvid >> 2;

  __shared__ float part[8 * 64];
  __shared__ float rowstat[64];

  Frag16 qf[2];
  #pragma unroll
  for (int kc = 0; kc < 2; ++kc)
    #pragma unroll
    for (int v = 0; v < 8; ++v) {
      int k0 = kpair(v, half);
      qf[kc].u[v] = *(const unsigned int*)
          &Qb[((size_t)(b * Nn + nbase + rt * 16 + l15)) * CQ + kc * 32 + k0];
    }

  for (int i = tid; i < 8 * 64; i += 256) part[i] = -3.0e38f;
  __syncthreads();

  // ---------------- pass 1: row max ----------------
  float rmax[8];
  #pragma unroll
  for (int j = 0; j < 8; ++j) rmax[j] = -3.0e38f;

  for (int mb = 0; mb < Nn; mb += 64) {
    __builtin_prefetch(&Kb[((size_t)(b * Nn + mb + 64)) * CQ + lane * 16], 0, 0);
    #pragma unroll
    for (int t = 0; t < 2; ++t) {
      int mcol = mb + (cp * 2 + t) * 16 + l15;
      v8f s = {};
      #pragma unroll
      for (int kc = 0; kc < 2; ++kc) {
        Frag16 kf;
        #pragma unroll
        for (int v = 0; v < 8; ++v) {
          int k0 = kpair(v, half);
          kf.u[v] = *(const unsigned int*)
              &Kb[((size_t)(b * Nn + mcol)) * CQ + kc * 32 + k0];
        }
        s = __builtin_amdgcn_wmma_f32_16x16x32_f16(
            false, qf[kc].h, false, kf.h, (short)0, s, false, false);
      }
      #pragma unroll
      for (int j = 0; j < 8; ++j) rmax[j] = fmaxf(rmax[j], s[j]);
    }
  }
  #pragma unroll
  for (int j = 0; j < 8; ++j) {
    float v = rmax[j];
    for (int m = 8; m >= 1; m >>= 1) v = fmaxf(v, __shfl_xor(v, m, 16));
    if (l15 == 0) part[wvid * 64 + rt * 16 + j + 8 * half] = v;
  }
  __syncthreads();
  if (tid < 64) {
    float v = part[tid];
    #pragma unroll
    for (int w = 1; w < 8; ++w) v = fmaxf(v, part[w * 64 + tid]);
    rowstat[tid] = v;
    gmax[(size_t)b * Nn + nbase + tid] = v;
  }
  __syncthreads();

  float mymax[8];
  #pragma unroll
  for (int j = 0; j < 8; ++j) mymax[j] = rowstat[rt * 16 + j + 8 * half];

  for (int i = tid; i < 8 * 64; i += 256) part[i] = 0.0f;
  __syncthreads();

  // ---------------- pass 2: row sum of exp ----------------
  float rsum[8];
  #pragma unroll
  for (int j = 0; j < 8; ++j) rsum[j] = 0.0f;

  for (int mb = 0; mb < Nn; mb += 64) {
    #pragma unroll
    for (int t = 0; t < 2; ++t) {
      int mcol = mb + (cp * 2 + t) * 16 + l15;
      v8f s = {};
      #pragma unroll
      for (int kc = 0; kc < 2; ++kc) {
        Frag16 kf;
        #pragma unroll
        for (int v = 0; v < 8; ++v) {
          int k0 = kpair(v, half);
          kf.u[v] = *(const unsigned int*)
              &Kb[((size_t)(b * Nn + mcol)) * CQ + kc * 32 + k0];
        }
        s = __builtin_amdgcn_wmma_f32_16x16x32_f16(
            false, qf[kc].h, false, kf.h, (short)0, s, false, false);
      }
      #pragma unroll
      for (int j = 0; j < 8; ++j) rsum[j] += __expf(s[j] - mymax[j]);
    }
  }
  #pragma unroll
  for (int j = 0; j < 8; ++j) {
    float v = rsum[j];
    for (int m = 8; m >= 1; m >>= 1) v += __shfl_xor(v, m, 16);
    if (l15 == 0) part[wvid * 64 + rt * 16 + j + 8 * half] = v;
  }
  __syncthreads();
  if (tid < 64) {
    float v = part[tid];
    #pragma unroll
    for (int w = 1; w < 8; ++w) v += part[w * 64 + tid];
    gsum[(size_t)b * Nn + nbase + tid] = v;
  }
}

// ---------------------------------------------------------------------------
// Kernel 3: O = softmax(Q K^T) V with fused epilogue gamma*O + x.
// Block: 64 query rows x 256 V channels. Per 64-key chunk:
//   - V tile (256 ch x 64 keys, f16) is DMA'd into LDS by the Tensor Data
//     Mover, one descriptor per wave (its own 32-row slice), double-buffered
//     so chunk i+1's TDM overlaps chunk i's WMMA work (s_wait_tensorcnt 1).
//   - 8 waves jointly WMMA the 64x64 S tile, exp() into an LDS f16 P tile.
//   - each wave runs P*V WMMAs for its 32 V channels (64 f32 accumulators per lane).
// ---------------------------------------------------------------------------
__global__ __launch_bounds__(256)
void attn_out_kernel(const _Float16* __restrict__ Qb, const _Float16* __restrict__ Kb,
                     const _Float16* __restrict__ Vt,
                     const float* __restrict__ gmax, const float* __restrict__ gsum,
                     const float* __restrict__ x, const float* __restrict__ gamma,
                     float* __restrict__ out)
{
  const int cg = blockIdx.x;              // 0..1 : 256 V channels
  const int nt = blockIdx.y, b = blockIdx.z;
  const int tid = threadIdx.x, lane = tid & 31, wvid = tid >> 5;
  const int half = lane >> 4, l15 = lane & 15;
  const int nbase = nt * 64;
  const int rt = wvid & 3, cp = wvid >> 2;
  const int cwave = cg * 256 + wvid * 32;

  __shared__ _Float16 sp[64 * 64];        // P tile [n][m]
  __shared__ _Float16 sv[2][256 * 64];    // V tiles [c_local][m], double-buffered

  // Q fragments for this wave's S row-subtile
  Frag16 qf[2];
  #pragma unroll
  for (int kc = 0; kc < 2; ++kc)
    #pragma unroll
    for (int v = 0; v < 8; ++v) {
      int k0 = kpair(v, half);
      qf[kc].u[v] = *(const unsigned int*)
          &Qb[((size_t)(b * Nn + nbase + rt * 16 + l15)) * CQ + kc * 32 + k0];
    }

  float mymax[8];
  #pragma unroll
  for (int j = 0; j < 8; ++j)
    mymax[j] = gmax[(size_t)b * Nn + nbase + rt * 16 + j + 8 * half];
  float invs[4][8];
  #pragma unroll
  for (int rr = 0; rr < 4; ++rr)
    #pragma unroll
    for (int j = 0; j < 8; ++j)
      invs[rr][j] = 1.0f / gsum[(size_t)b * Nn + nbase + rr * 16 + j + 8 * half];

  v8f acc[4][2] = {};

  // TDM setup: this wave's 32-channel slice of Vt
  const _Float16* vsrc = Vt + ((size_t)(b * Cc + cwave)) * Nn;
  const unsigned ldsw[2] = {
    (unsigned)(size_t)&sv[0][wvid * 32 * 64],
    (unsigned)(size_t)&sv[1][wvid * 32 * 64]
  };
  // prologue: DMA chunk 0 into buffer 0
  tdm_load_2d_f16(ldsw[0], vsrc, /*tile*/64, 32, /*tensor*/Nn, Cc, /*stride*/Nn);

  for (int i = 0; i < NCH; ++i) {
    const int mb = i * 64;
    const bool more = (i + 1 < NCH);
    if (more)   // DMA next chunk into the other buffer (overlaps compute)
      tdm_load_2d_f16(ldsw[(i + 1) & 1], vsrc + (mb + 64), 64, 32, Nn, Cc, Nn);

    // ---- S phase: this wave's 2 tiles of the 64x64 S chunk -> exp -> LDS ----
    __builtin_prefetch(&Kb[((size_t)(b * Nn + mb + 64)) * CQ + lane * 16], 0, 0);
    #pragma unroll
    for (int t = 0; t < 2; ++t) {
      int ct = cp * 2 + t;
      int mcol = mb + ct * 16 + l15;
      v8f s = {};
      #pragma unroll
      for (int kc = 0; kc < 2; ++kc) {
        Frag16 kf;
        #pragma unroll
        for (int v = 0; v < 8; ++v) {
          int k0 = kpair(v, half);
          kf.u[v] = *(const unsigned int*)
              &Kb[((size_t)(b * Nn + mcol)) * CQ + kc * 32 + k0];
        }
        s = __builtin_amdgcn_wmma_f32_16x16x32_f16(
            false, qf[kc].h, false, kf.h, (short)0, s, false, false);
      }
      #pragma unroll
      for (int j = 0; j < 8; ++j) {
        float p = __expf(s[j] - mymax[j]);
        sp[(rt * 16 + j + 8 * half) * 64 + ct * 16 + l15] = (_Float16)p;
      }
    }

    // current chunk's V tile must have landed (1 TDM may remain in flight)
    wait_tensorcnt_le(more ? 1 : 0);
    __syncthreads();

    // ---- O phase: acc += P(64xk) * V(kx32) from LDS ----
    const _Float16* svc = &sv[i & 1][0];
    #pragma unroll
    for (int kc = 0; kc < 2; ++kc) {
      Frag16 bfr[2];
      #pragma unroll
      for (int t = 0; t < 2; ++t) {
        int cl = wvid * 32 + t * 16 + l15;
        #pragma unroll
        for (int v = 0; v < 8; ++v) {
          int k0 = kpair(v, half);
          bfr[t].u[v] = *(const unsigned int*)&svc[cl * 64 + kc * 32 + k0];
        }
      }
      #pragma unroll
      for (int rr = 0; rr < 4; ++rr) {
        Frag16 af;
        #pragma unroll
        for (int v = 0; v < 8; ++v) {
          int k0 = kpair(v, half);
          af.u[v] = *(const unsigned int*)&sp[(rr * 16 + l15) * 64 + kc * 32 + k0];
        }
        #pragma unroll
        for (int t = 0; t < 2; ++t)
          acc[rr][t] = __builtin_amdgcn_wmma_f32_16x16x32_f16(
              false, af.h, false, bfr[t].h, (short)0, acc[rr][t], false, false);
      }
    }
    __syncthreads();
  }

  // epilogue: out[b,c,n] = gamma * O[n,c]/sum[n] + x[b,c,n]
  const float g = gamma[0];
  #pragma unroll
  for (int rr = 0; rr < 4; ++rr)
    #pragma unroll
    for (int t = 0; t < 2; ++t)
      #pragma unroll
      for (int j = 0; j < 8; ++j) {
        int n = nbase + rr * 16 + j + 8 * half;
        int c = cwave + t * 16 + l15;
        size_t idx = ((size_t)(b * Cc + c)) * Nn + n;
        out[idx] = g * (acc[rr][t][j] * invs[rr][j]) + x[idx];
      }
}

// ---------------------------------------------------------------------------
extern "C" void kernel_launch(void* const* d_in, const int* in_sizes, int n_in,
                              void* d_out, int out_size, void* d_ws, size_t ws_size,
                              hipStream_t stream) {
  (void)in_sizes; (void)n_in; (void)out_size; (void)ws_size;
  const float* x     = (const float*)d_in[0];
  const float* wq    = (const float*)d_in[1];
  const float* bq    = (const float*)d_in[2];
  const float* wk    = (const float*)d_in[3];
  const float* bk    = (const float*)d_in[4];
  const float* wvm   = (const float*)d_in[5];
  const float* bv    = (const float*)d_in[6];
  const float* gamma = (const float*)d_in[7];
  float* out = (float*)d_out;

  char* ws = (char*)d_ws;
  _Float16* Qb  = (_Float16*)(ws);                                //  2 MB
  _Float16* Kb  = (_Float16*)(ws + ((size_t)2 << 20));            //  2 MB
  _Float16* Vt  = (_Float16*)(ws + ((size_t)4 << 20));            // 16 MB
  float*   gmax = (float*)(ws + ((size_t)20 << 20));              // 64 KB
  float*   gsum = (float*)(ws + ((size_t)20 << 20) + (64 << 10)); // 64 KB
  _Float16* Wh  = (_Float16*)(ws + ((size_t)21 << 20));           // 640 KB
  float*  biasc = (float*)(ws + ((size_t)22 << 20));              // 2.5 KB

  prep_w_kernel<<<dim3(1280), 256, 0, stream>>>(wq, bq, wk, bk, wvm, bv, Wh, biasc);
  proj_kernel<<<dim3(5, 64, 4), 256, 0, stream>>>(x, Wh, biasc, Qb, Kb, Vt);
  stats_kernel<<<dim3(64, 4), 256, 0, stream>>>(Qb, Kb, gmax, gsum);
  attn_out_kernel<<<dim3(2, 64, 4), 256, 0, stream>>>(Qb, Kb, Vt, gmax, gsum,
                                                      x, gamma, out);
}
